// AnimaPHCorrected_85401129713705
// MI455X (gfx1250) — matrix operand order repaired and checked by reading, not compiled, not run on hardware
//
#include <hip/hip_runtime.h>
#include <math.h>

// ---------------------------------------------------------------------------
// Problem constants (match the JAX reference)
// ---------------------------------------------------------------------------
#define B_ROWS   4096
#define D_IN     1024
#define D_H      4096
#define D_OUT    1024
#define NEXP     8
#define N_ACTIVE 5
#define N_CAMP_A 4
#define TEMP_F   2.718281828459045f
#define NTOT     (NEXP * D_H)        // 32768: stacked expert hidden dim
#define KC       (N_CAMP_A * D_H)    // 16384: per-camp K for GEMM2

// WMMA tile config
#define BT_M 128
#define BT_N 128
#define KSTAGE 64                    // K per LDS stage (2 WMMA k-substeps)
#define KSTR 72                      // LDS row stride (bf16): 144B rows, b128-aligned

// --------------------------- feature detection -----------------------------
#if defined(__has_builtin)
#  if __has_builtin(__builtin_amdgcn_global_load_async_to_lds_b128)
#    define HAVE_ASYNC_LDS 1
#  endif
#  if __has_builtin(__builtin_amdgcn_s_wait_asynccnt)
#    define HAVE_WAIT_ASYNC_BUILTIN 1
#  endif
#endif
#ifndef HAVE_ASYNC_LDS
#  define HAVE_ASYNC_LDS 0
#endif
#ifndef HAVE_WAIT_ASYNC_BUILTIN
#  define HAVE_WAIT_ASYNC_BUILTIN 0
#endif

#if HAVE_WAIT_ASYNC_BUILTIN
#  define WAIT_ASYNCCNT(n) __builtin_amdgcn_s_wait_asynccnt(n)
#else
#  define WAIT_ASYNCCNT(n) asm volatile("s_wait_asynccnt %0" :: "n"(n) : "memory")
#endif

typedef __attribute__((ext_vector_type(16))) __bf16 bf16x16;
typedef __attribute__((ext_vector_type(8)))  __bf16 bf16x8;
typedef __attribute__((ext_vector_type(8)))  float  f32x8;
typedef __attribute__((ext_vector_type(4)))  int    i32x4;

typedef __attribute__((address_space(1))) i32x4* gvec_ptr;   // global int4*
typedef __attribute__((address_space(3))) i32x4* lvec_ptr;   // LDS int4*

union BF16Frag { bf16x16 v; bf16x8 h[2]; };

__device__ __forceinline__ unsigned short f32_to_bf16(float f) {
    unsigned int u = __float_as_uint(f);
    unsigned int lsb = (u >> 16) & 1u;
    u += 0x7fffu + lsb;              // round-to-nearest-even
    return (unsigned short)(u >> 16);
}

// 16-byte global -> LDS copy: async (ASYNCcnt-tracked DMA) when available.
__device__ __forceinline__ void copy16_g2l(const __bf16* __restrict__ g,
                                           __bf16* __restrict__ l) {
#if HAVE_ASYNC_LDS
    // builtin signature (from probe diagnostic): (AS1 int4*, AS3 int4*, i32, i32)
    gvec_ptr ga = (gvec_ptr)g;
    // flat LDS address low 32 bits == LDS byte offset (aperture rule)
    lvec_ptr la = (lvec_ptr)(unsigned int)(unsigned long long)l;
    __builtin_amdgcn_global_load_async_to_lds_b128(ga, la, 0, 0);
#else
    *(bf16x8*)l = *(const bf16x8*)g;
#endif
}

__device__ __forceinline__ void wait_stage(bool more_inflight) {
#if HAVE_ASYNC_LDS
    if (more_inflight) WAIT_ASYNCCNT(8);   // 8 asyncs of the *next* stage remain
    else               WAIT_ASYNCCNT(0);
#else
    (void)more_inflight;
#endif
}

// ---------------------------------------------------------------------------
// 1) f32 -> bf16 convert (x)
// ---------------------------------------------------------------------------
__global__ void cvt_f32_bf16_kernel(const float* __restrict__ in,
                                    unsigned short* __restrict__ out, int n) {
    int i = (blockIdx.x * blockDim.x + threadIdx.x) * 4;
    if (i + 3 < n) {
        float4 v = *(const float4*)(in + i);
        out[i + 0] = f32_to_bf16(v.x);
        out[i + 1] = f32_to_bf16(v.y);
        out[i + 2] = f32_to_bf16(v.z);
        out[i + 3] = f32_to_bf16(v.w);
    }
}

// ---------------------------------------------------------------------------
// 2) Tiled transpose + f32->bf16 : in [z][R][C] f32 -> out [z][C][R] bf16
// ---------------------------------------------------------------------------
__global__ void transpose_cvt_kernel(const float* __restrict__ in,
                                     unsigned short* __restrict__ out,
                                     int R, int C) {
    __shared__ float tile[32][33];
    const float* src = in + (size_t)blockIdx.z * R * C;
    unsigned short* dst = out + (size_t)blockIdx.z * R * C;
    int c0 = blockIdx.x * 32, r0 = blockIdx.y * 32;
    int x = threadIdx.x, y = threadIdx.y;           // 32 x 8
    #pragma unroll
    for (int j = 0; j < 32; j += 8)
        tile[y + j][x] = src[(size_t)(r0 + y + j) * C + c0 + x];
    __syncthreads();
    #pragma unroll
    for (int j = 0; j < 32; j += 8)
        dst[(size_t)(c0 + y + j) * R + r0 + x] = f32_to_bf16(tile[x][y + j]);
}

// ---------------------------------------------------------------------------
// 3) Boltzmann gate: softmax(T=e) -> top-5 mask -> renorm weights
// ---------------------------------------------------------------------------
__global__ void gate_kernel(const float* __restrict__ x,
                            const float* __restrict__ gate_w,   // [D_IN][E]
                            const float* __restrict__ gate_b,
                            float* __restrict__ gw_out) {       // [B][E]
    __shared__ float red[256][NEXP];
    int b = blockIdx.x, tid = threadIdx.x;
    float acc[NEXP];
    #pragma unroll
    for (int e = 0; e < NEXP; ++e) acc[e] = 0.f;
    for (int i = tid; i < D_IN; i += 256) {
        float xv = x[(size_t)b * D_IN + i];
        #pragma unroll
        for (int e = 0; e < NEXP; ++e) acc[e] += xv * gate_w[i * NEXP + e];
    }
    #pragma unroll
    for (int e = 0; e < NEXP; ++e) red[tid][e] = acc[e];
    __syncthreads();
    for (int s = 128; s > 0; s >>= 1) {
        if (tid < s) {
            #pragma unroll
            for (int e = 0; e < NEXP; ++e) red[tid][e] += red[tid + s][e];
        }
        __syncthreads();
    }
    if (tid == 0) {
        float sc[NEXP], p[NEXP];
        float mx = -1e30f;
        #pragma unroll
        for (int e = 0; e < NEXP; ++e) {
            sc[e] = (red[0][e] + gate_b[e]) * (1.0f / TEMP_F);
            mx = fmaxf(mx, sc[e]);
        }
        float sum = 0.f;
        #pragma unroll
        for (int e = 0; e < NEXP; ++e) { p[e] = expf(sc[e] - mx); sum += p[e]; }
        #pragma unroll
        for (int e = 0; e < NEXP; ++e) p[e] /= sum;
        bool sel[NEXP];
        #pragma unroll
        for (int e = 0; e < NEXP; ++e) sel[e] = false;
        for (int t = 0; t < N_ACTIVE; ++t) {
            int best = -1; float bv = -1.f;
            #pragma unroll
            for (int e = 0; e < NEXP; ++e)
                if (!sel[e] && p[e] > bv) { bv = p[e]; best = e; }
            sel[best] = true;
        }
        float ws = 0.f;
        #pragma unroll
        for (int e = 0; e < NEXP; ++e) if (sel[e]) ws += p[e];
        float inv = 1.f / (ws + 1e-8f);
        #pragma unroll
        for (int e = 0; e < NEXP; ++e)
            gw_out[(size_t)b * NEXP + e] = sel[e] ? p[e] * inv : 0.f;
    }
}

// ---------------------------------------------------------------------------
// WMMA mainloop pieces: A [M,K] bf16, BT [N,K] bf16, staged KSTAGE=64 at a time
// ---------------------------------------------------------------------------
__device__ __forceinline__ void stage_issue(
    __bf16 (*As)[KSTR], __bf16 (*Bs)[KSTR],
    const __bf16* __restrict__ A, const __bf16* __restrict__ BT,
    size_t lda, size_t ldb, int m0, int n0, size_t k0, int tid) {
    // 128 rows x 64 bf16 per tile => 1024 chunks of 8 bf16 (16B), 4 per thread
    #pragma unroll
    for (int i = 0; i < 4; ++i) {
        int c = i * 256 + tid;
        int r = c >> 3, cc = (c & 7) << 3;
        copy16_g2l(A  + (size_t)(m0 + r) * lda + k0 + cc, &As[r][cc]);
        copy16_g2l(BT + (size_t)(n0 + r) * ldb + k0 + cc, &Bs[r][cc]);
    }
}

__device__ __forceinline__ void wmma_ksub(
    const __bf16 (*As)[KSTR], const __bf16 (*Bs)[KSTR], f32x8* acc,
    int wave, int lane, int ksub) {
    const int mrow = wave * 16 + (lane & 15);
    const int koff = ksub + ((lane >> 4) << 3);   // A lanes 16-31: K=8..15/24..31
    BF16Frag af;
    af.h[0] = *(const bf16x8*)&As[mrow][koff];
    af.h[1] = *(const bf16x8*)&As[mrow][koff + 16];
    const int kb = ksub + ((lane >> 4) << 3);     // B lanes 16-31: K=8..15/24..31
    #pragma unroll
    for (int half = 0; half < 2; ++half) {
        BF16Frag bfv[4];
        #pragma unroll
        for (int q = 0; q < 4; ++q) {             // gather 4 B frags, one wait
            int brow = (half * 4 + q) * 16 + (lane & 15);
            bfv[q].h[0] = *(const bf16x8*)&Bs[brow][kb];
            bfv[q].h[1] = *(const bf16x8*)&Bs[brow][kb + 16];
        }
        #pragma unroll
        for (int q = 0; q < 4; ++q)               // 4 back-to-back WMMAs
            acc[half * 4 + q] = __builtin_amdgcn_wmma_f32_16x16x32_bf16(
                false, af.v, false, bfv[q].v, (short)0, acc[half * 4 + q],
                false, false);
    }
}

// ---------------------------------------------------------------------------
// 4) GEMM1: H[b, e*H+j] = relu(x@w1 + b1) * gate_weight[b,e]  (bf16 out)
//    A = xb [B][D_IN], BT = w1t [NTOT][D_IN]
// ---------------------------------------------------------------------------
__global__ __launch_bounds__(256, 2)
void gemm1_kernel(const __bf16* __restrict__ A, const __bf16* __restrict__ BT,
                  const float* __restrict__ b1, const float* __restrict__ gw,
                  unsigned short* __restrict__ H) {
    __shared__ __bf16 As[2][BT_M][KSTR];
    __shared__ __bf16 Bs[2][BT_N][KSTR];
    const int n0 = blockIdx.x * BT_N;
    const int m0 = blockIdx.y * BT_M;
    const int tid = threadIdx.x, wave = tid >> 5, lane = tid & 31;
    f32x8 acc[8];
    #pragma unroll
    for (int i = 0; i < 8; ++i)
        #pragma unroll
        for (int j = 0; j < 8; ++j) acc[i][j] = 0.f;

    const int NST = D_IN / KSTAGE;   // 16
    stage_issue(As[0], Bs[0], A, BT, D_IN, D_IN, m0, n0, 0, tid);
    for (int s = 0; s < NST; ++s) {
        const bool more = (s + 1 < NST);
        if (more)
            stage_issue(As[(s + 1) & 1], Bs[(s + 1) & 1], A, BT, D_IN, D_IN,
                        m0, n0, (size_t)(s + 1) * KSTAGE, tid);
        wait_stage(more);
        __syncthreads();
        wmma_ksub(As[s & 1], Bs[s & 1], acc, wave, lane, 0);
        wmma_ksub(As[s & 1], Bs[s & 1], acc, wave, lane, 32);
        __syncthreads();
    }

    // epilogue: bias + relu + gate-weight scale -> bf16
    const int e = n0 >> 12;                     // expert id (block-uniform)
    float wrow[8];
    #pragma unroll
    for (int g = 0; g < 8; ++g) {
        int rowg = m0 + wave * 16 + ((lane >> 4) << 3) + g;
        wrow[g] = gw[(size_t)rowg * NEXP + e];
    }
    #pragma unroll
    for (int nn = 0; nn < 8; ++nn) {
        int coln = n0 + nn * 16 + (lane & 15);
        float bias = b1[(e << 12) + (coln & (D_H - 1))];
        #pragma unroll
        for (int g = 0; g < 8; ++g) {
            int rowg = m0 + wave * 16 + ((lane >> 4) << 3) + g;
            float v = fmaxf(acc[nn][g] + bias, 0.f) * wrow[g];
            H[(size_t)rowg * NTOT + coln] = f32_to_bf16(v);
        }
    }
}

// ---------------------------------------------------------------------------
// 5) GEMM2: camp-summed expert outputs.
//    A = H [B][NTOT] (camp slice), BT = w2t [D_OUT][NTOT]; z=0 -> RA, z=1 -> RG
// ---------------------------------------------------------------------------
__global__ __launch_bounds__(256, 2)
void gemm2_kernel(const __bf16* __restrict__ H, const __bf16* __restrict__ W2T,
                  float* __restrict__ RA, float* __restrict__ RG) {
    __shared__ __bf16 As[2][BT_M][KSTR];
    __shared__ __bf16 Bs[2][BT_N][KSTR];
    const int n0 = blockIdx.x * BT_N;
    const int m0 = blockIdx.y * BT_M;
    const int camp = blockIdx.z;
    const size_t koffs = (size_t)camp * KC;
    const int tid = threadIdx.x, wave = tid >> 5, lane = tid & 31;
    f32x8 acc[8];
    #pragma unroll
    for (int i = 0; i < 8; ++i)
        #pragma unroll
        for (int j = 0; j < 8; ++j) acc[i][j] = 0.f;

    const int NST = KC / KSTAGE;     // 256
    stage_issue(As[0], Bs[0], H, W2T, NTOT, NTOT, m0, n0, koffs, tid);
    for (int s = 0; s < NST; ++s) {
        const bool more = (s + 1 < NST);
        if (more)
            stage_issue(As[(s + 1) & 1], Bs[(s + 1) & 1], H, W2T, NTOT, NTOT,
                        m0, n0, koffs + (size_t)(s + 1) * KSTAGE, tid);
        wait_stage(more);
        __syncthreads();
        wmma_ksub(As[s & 1], Bs[s & 1], acc, wave, lane, 0);
        wmma_ksub(As[s & 1], Bs[s & 1], acc, wave, lane, 32);
        __syncthreads();
    }

    float* C = camp ? RG : RA;
    #pragma unroll
    for (int nn = 0; nn < 8; ++nn) {
        int coln = n0 + nn * 16 + (lane & 15);
        #pragma unroll
        for (int g = 0; g < 8; ++g) {
            int rowg = m0 + wave * 16 + ((lane >> 4) << 3) + g;
            C[(size_t)rowg * D_OUT + coln] = acc[nn][g];
        }
    }
}

// ---------------------------------------------------------------------------
// 6) Finalize: add Σ w_e*b2 bias, repulsion, PH correction, write 3 outputs
// ---------------------------------------------------------------------------
__global__ void finalize_kernel(const float* __restrict__ RA,
                                const float* __restrict__ RG,
                                const float* __restrict__ b2,   // [E][D_OUT]
                                const float* __restrict__ gw,   // [B][E]
                                const float* __restrict__ alpha,
                                const float* __restrict__ beta,
                                float* __restrict__ out) {
    __shared__ float rs[256], rss[256];
    __shared__ float wloc[NEXP];
    __shared__ float corr_sh;
    const int b = blockIdx.x, tid = threadIdx.x;
    if (tid < NEXP) wloc[tid] = gw[(size_t)b * NEXP + tid];
    __syncthreads();

    float av[4], gv[4], rv[4];
    float s1 = 0.f, s2 = 0.f;
    #pragma unroll
    for (int j = 0; j < 4; ++j) {
        int o = tid + j * 256;
        float a = RA[(size_t)b * D_OUT + o];
        float g = RG[(size_t)b * D_OUT + o];
        #pragma unroll
        for (int e = 0; e < N_CAMP_A; ++e) a += wloc[e] * b2[e * D_OUT + o];
        #pragma unroll
        for (int e = N_CAMP_A; e < NEXP; ++e) g += wloc[e] * b2[e * D_OUT + o];
        av[j] = a; gv[j] = g;
        float r = a - g;
        rv[j] = r;
        s1 += r; s2 += r * r;
    }
    rs[tid] = s1; rss[tid] = s2;
    __syncthreads();
    for (int s = 128; s > 0; s >>= 1) {
        if (tid < s) { rs[tid] += rs[tid + s]; rss[tid] += rss[tid + s]; }
        __syncthreads();
    }
    if (tid == 0) {
        float S1 = rs[0], S2 = rss[0];
        float l2 = sqrtf(S2);
        float mean = S1 * (1.0f / D_OUT);
        float var = S2 * (1.0f / D_OUT) - mean * mean;
        float ph = l2 * (1.0f + var);
        float z = alpha[0] * ph + beta[0];
        corr_sh = 2.0f / (1.0f + expf(-z));
    }
    __syncthreads();
    const float corr = corr_sh;
    const size_t BO = (size_t)B_ROWS * D_OUT;
    #pragma unroll
    for (int j = 0; j < 4; ++j) {
        int o = tid + j * 256;
        size_t idx = (size_t)b * D_OUT + o;
        out[idx]          = rv[j] * corr;   // output
        out[BO + idx]     = av[j];          // out_a
        out[2 * BO + idx] = gv[j];          // out_g
    }
}

// ---------------------------------------------------------------------------
// Launch
// ---------------------------------------------------------------------------
extern "C" void kernel_launch(void* const* d_in, const int* in_sizes, int n_in,
                              void* d_out, int out_size, void* d_ws, size_t ws_size,
                              hipStream_t stream) {
    const float* x      = (const float*)d_in[0];
    const float* gate_w = (const float*)d_in[1];
    const float* gate_b = (const float*)d_in[2];
    const float* w1     = (const float*)d_in[3];
    const float* b1     = (const float*)d_in[4];
    const float* w2     = (const float*)d_in[5];
    const float* b2     = (const float*)d_in[6];
    const float* ph_a   = (const float*)d_in[7];
    const float* ph_b   = (const float*)d_in[8];
    float* out = (float*)d_out;

    // workspace layout (256B aligned)
    char* ws = (char*)d_ws;
    size_t off = 0;
    auto take = [&](size_t bytes) { char* p = ws + off;
        off += (bytes + 255) & ~(size_t)255; return p; };
    unsigned short* xb  = (unsigned short*)take((size_t)B_ROWS * D_IN * 2);
    unsigned short* w1t = (unsigned short*)take((size_t)NEXP * D_IN * D_H * 2);
    unsigned short* w2t = (unsigned short*)take((size_t)NEXP * D_H * D_OUT * 2);
    float*          gw  = (float*)take((size_t)B_ROWS * NEXP * 4);
    unsigned short* h   = (unsigned short*)take((size_t)B_ROWS * NTOT * 2);
    float*          ra  = (float*)take((size_t)B_ROWS * D_OUT * 4);
    float*          rg  = (float*)take((size_t)B_ROWS * D_OUT * 4);

    // 1) convert x -> bf16
    {
        int n = B_ROWS * D_IN;
        cvt_f32_bf16_kernel<<<n / (256 * 4), 256, 0, stream>>>(x, xb, n);
    }
    // 2) transpose+convert weights: w1 [e][D_IN][D_H] -> w1t [e][D_H][D_IN]
    transpose_cvt_kernel<<<dim3(D_H / 32, D_IN / 32, NEXP), dim3(32, 8), 0, stream>>>(
        w1, w1t, D_IN, D_H);
    //    w2 stacked [NTOT][D_OUT] -> w2t [D_OUT][NTOT]
    transpose_cvt_kernel<<<dim3(D_OUT / 32, NTOT / 32, 1), dim3(32, 8), 0, stream>>>(
        w2, w2t, NTOT, D_OUT);
    // 3) gate weights
    gate_kernel<<<B_ROWS, 256, 0, stream>>>(x, gate_w, gate_b, gw);
    // 4) GEMM1 (fused bias+relu+gate scale)
    gemm1_kernel<<<dim3(NTOT / BT_N, B_ROWS / BT_M), 256, 0, stream>>>(
        (const __bf16*)xb, (const __bf16*)w1t, b1, gw, h);
    // 5) GEMM2 (both camps)
    gemm2_kernel<<<dim3(D_OUT / BT_N, B_ROWS / BT_M, 2), 256, 0, stream>>>(
        (const __bf16*)h, (const __bf16*)w2t, ra, rg);
    // 6) finalize
    finalize_kernel<<<B_ROWS, 256, 0, stream>>>(ra, rg, b2, gw, ph_a, ph_b, out);
}
